// FourierEFA_MACE_81716047774379
// MI455X (gfx1250) — matrix operand order repaired and checked by reading, not compiled, not run on hardware
//
#include <hip/hip_runtime.h>
#include <math.h>

typedef __attribute__((ext_vector_type(16))) _Float16 v16h;
typedef __attribute__((ext_vector_type(8)))  _Float16 v8h;
typedef __attribute__((ext_vector_type(8)))  float    v8f;
typedef __attribute__((ext_vector_type(4)))  float    v4f;

#define HDIM  64
#define MODES 6

__device__ __forceinline__ float silu_f(float x) { return x / (1.0f + __expf(-x)); }

__device__ __forceinline__ v8f wmma16(v16h A, v16h B, v8f C) {
    // D = A(16x32 f16) x B(32x16 f16) + C(16x16 f32)
    return __builtin_amdgcn_wmma_f32_16x16x32_f16(false, A, false, B, (short)0, C, false, false);
}

// ---------------------------------------------------------------------------
// Kernel 1: per-graph box -> top-6 smallest |k|^2 Fourier vectors; node ranges.
// ---------------------------------------------------------------------------
__global__ void fefa_setup_kernel(const float* __restrict__ cell,
                                  const int*   __restrict__ batch,
                                  int N, int G,
                                  float* __restrict__ kvec,   // [G][MODES][3]
                                  int*   __restrict__ range)  // [G+1]
{
    const int t = threadIdx.x;
    const float TWO_PI = 6.28318530717958647692f;
    if (t < G) {
        const int g = t;
        float ix = TWO_PI / cell[g * 9 + 0];
        float iy = TWO_PI / cell[g * 9 + 4];
        float iz = TWO_PI / cell[g * 9 + 8];
        float k2max = (TWO_PI * 0.1f) * (TWO_PI * 0.1f);   // (2pi/DL)^2, DL=10
        int chosen[MODES];
        for (int sel = 0; sel < MODES; ++sel) {
            float best = 3.0e38f; int bj = 0;
            for (int j = 0; j < 342; ++j) {
                bool used = false;
                for (int p = 0; p < sel; ++p) used |= (chosen[p] == j);
                if (used) continue;
                int jj = (j < 171) ? j : j + 1;            // skip (0,0,0)
                int a = jj / 49, b = (jj / 7) % 7, c = jj % 7;
                float kx = (float)(a - 3) * ix;
                float ky = (float)(b - 3) * iy;
                float kz = (float)(c - 3) * iz;
                float ksq = kx * kx + ky * ky + kz * kz;
                float val = (ksq <= k2max) ? ksq : 1.0e10f; // mirror reference masking
                if (val < best) { best = val; bj = j; }     // strict <: stable ties
            }
            chosen[sel] = bj;
            int jj = (bj < 171) ? bj : bj + 1;
            int a = jj / 49, b = (jj / 7) % 7, c = jj % 7;
            kvec[(g * MODES + sel) * 3 + 0] = (float)(a - 3) * ix;
            kvec[(g * MODES + sel) * 3 + 1] = (float)(b - 3) * iy;
            kvec[(g * MODES + sel) * 3 + 2] = (float)(c - 3) * iz;
        }
    } else if (t >= 64 && t < 64 + G + 1) {
        const int g = t - 64;                   // lower_bound(batch, g)
        int lo = 0, hi = N;
        while (lo < hi) { int mid = (lo + hi) >> 1; if (batch[mid] < g) lo = mid + 1; else hi = mid; }
        range[g] = lo;                          // range[G] == N
    }
}

// ---------------------------------------------------------------------------
// Kernel 2: qkv = node_feat @ W_qkv^T via WMMA f16; silu(q), silu(k); f32 out.
// One block = 16 node rows, 4 waves x 3 col-tiles of 16 (192 cols).
// ---------------------------------------------------------------------------
__global__ __launch_bounds__(128) void fefa_qkv_kernel(
    const float* __restrict__ nf, const float* __restrict__ W,
    float* __restrict__ qs, float* __restrict__ ks, float* __restrict__ vs, int N)
{
    const int lane = threadIdx.x & 31;
    const int wid  = threadIdx.x >> 5;      // 0..3
    const int col  = lane & 15;
    const int half = lane >> 4;
    const int rowbase = blockIdx.x * 16;

    v8f acc[3];
#pragma unroll
    for (int j = 0; j < 3; ++j)
#pragma unroll
        for (int r = 0; r < 8; ++r) acc[j][r] = 0.0f;

#pragma unroll
    for (int kk = 0; kk < 2; ++kk) {        // K = 64 in two 32-steps
        // A fragment: 16 node rows x 32 features
        int node = rowbase + col; if (node >= N) node = N - 1;
        const float* ap = nf + (size_t)node * HDIM + kk * 32;
        v16h A;
        {
            v4f f0 = *(const v4f*)(ap + half * 16);
            v4f f1 = *(const v4f*)(ap + half * 16 + 4);
            v4f f2 = *(const v4f*)(ap + half * 16 + 8);
            v4f f3 = *(const v4f*)(ap + half * 16 + 12);
#pragma unroll
            for (int i = 0; i < 4; ++i) {
                A[i]      = (_Float16)f0[i];  A[4 + i]  = (_Float16)f1[i];
                A[8 + i]  = (_Float16)f2[i];  A[12 + i] = (_Float16)f3[i];
            }
        }
#pragma unroll
        for (int j = 0; j < 3; ++j) {
            const int J = wid * 3 + j;      // 0..11
            const int o = J * 16 + col;     // output column (0..191)
            const float* bp = W + (size_t)o * HDIM + kk * 32 + half * 16;
            v16h B;
            v4f g0 = *(const v4f*)(bp + 0);
            v4f g1 = *(const v4f*)(bp + 4);
            v4f g2 = *(const v4f*)(bp + 8);
            v4f g3 = *(const v4f*)(bp + 12);
#pragma unroll
            for (int i = 0; i < 4; ++i) {
                B[i]      = (_Float16)g0[i];  B[4 + i]  = (_Float16)g1[i];
                B[8 + i]  = (_Float16)g2[i];  B[12 + i] = (_Float16)g3[i];
            }
            acc[j] = wmma16(A, B, acc[j]);
        }
    }

#pragma unroll
    for (int j = 0; j < 3; ++j) {
        const int J = wid * 3 + j;
        const int o = J * 16 + col;
        float* dst; int oc; bool act;
        if (J < 4)      { dst = qs; oc = o;       act = true;  }
        else if (J < 8) { dst = ks; oc = o - 64;  act = true;  }
        else            { dst = vs; oc = o - 128; act = false; }
#pragma unroll
        for (int r = 0; r < 8; ++r) {
            int node = rowbase + r + half * 8;
            float x = acc[j][r];
            if (act) x = silu_f(x);
            if (node < N) dst[(size_t)node * HDIM + oc] = x;
        }
    }
}

// ---------------------------------------------------------------------------
// Kernel 3: per-graph linear attention over 6 Fourier modes.
//   Pass 1: S_m = K_m^T V  (64x64, f32 acc kept in VGPRs across node chunks)
//   Pass 2: out = sum_m (Q_m/8) @ S_m
// 8 waves / block; 32-node LDS-staged chunks; f16 operands, f32 accumulate.
// Mode loops are FULLY UNROLLED so accS stays register-resident (no scratch).
// ---------------------------------------------------------------------------
__global__ __launch_bounds__(256) void fefa_attn_kernel(
    const float* __restrict__ qs, const float* __restrict__ ksl,
    const float* __restrict__ vs, const float* __restrict__ pos,
    const float* __restrict__ kvec, const int* __restrict__ range,
    float* __restrict__ out)
{
    const int g = blockIdx.x;
    const int s = range[g], e = range[g + 1];
    const int cnt = e - s;
    const int t = threadIdx.x;
    const int lane = t & 31, wid = t >> 5;
    const int col = lane & 15, half = lane >> 4;

    __shared__ __align__(16) _Float16 sS[MODES][HDIM * HDIM]; // S^T: [h2][h1]
    __shared__ __align__(16) _Float16 sA[HDIM * 32];          // K_r^T [h][nl]  (pass2: Q [nl][h])
    __shared__ __align__(16) _Float16 sB[HDIM * 32];          // V^T   [h][nl]

    const int nl = t & 31;         // staged node slot
    const int hb = t >> 5;         // staged h block (0..7)

    // hoist the 6 mode k-vectors into registers (uniform scalar loads)
    float kvx[MODES], kvy[MODES], kvz[MODES];
#pragma unroll
    for (int m = 0; m < MODES; ++m) {
        kvx[m] = kvec[(g * MODES + m) * 3 + 0];
        kvy[m] = kvec[(g * MODES + m) * 3 + 1];
        kvz[m] = kvec[(g * MODES + m) * 3 + 2];
    }

    v8f accS[MODES][2];
#pragma unroll
    for (int m = 0; m < MODES; ++m)
#pragma unroll
        for (int u = 0; u < 2; ++u)
#pragma unroll
            for (int r = 0; r < 8; ++r) accS[m][u][r] = 0.0f;

    // ---------------- Pass 1: S_m = K_m^T V ----------------
    for (int c = 0; c < cnt; c += 32) {
        __syncthreads();
        {   // stage V^T chunk (mode-independent)
            int n = s + c + nl; bool ok = (n < e);
            const float* vp = vs + (size_t)n * HDIM;
            __builtin_prefetch(vs + (size_t)(n + 32) * HDIM, 0, 1);
#pragma unroll
            for (int i = 0; i < 8; ++i) {
                int h = hb * 8 + i;
                sB[h * 32 + nl] = (_Float16)(ok ? vp[h] : 0.0f);
            }
        }
#pragma unroll
        for (int m = 0; m < MODES; ++m) {       // UNROLLED: accS stays in VGPRs
            if (m) __syncthreads();
            {   // stage roped K^T chunk for mode m
                int n = s + c + nl; bool ok = (n < e);
                float ph = 0.0f;
                if (ok) ph = pos[n * 3] * kvx[m] + pos[n * 3 + 1] * kvy[m] + pos[n * 3 + 2] * kvz[m];
                float cc = __cosf(ph), sn = __sinf(ph);
#pragma unroll
                for (int i = 0; i < 8; ++i) {
                    int h = hb * 8 + i;
                    float k1 = ok ? ksl[(size_t)(s + c + nl) * HDIM + h] : 0.0f;
                    float k2 = ok ? ksl[(size_t)(s + c + nl) * HDIM + (h ^ 32)] : 0.0f;
                    float val = k1 * cc + ((h < 32) ? -k2 * sn : k2 * sn);
                    sA[h * 32 + nl] = (_Float16)val;
                }
            }
            __syncthreads();
#pragma unroll
            for (int u = 0; u < 2; ++u) {      // each wave: 2 of 16 output tiles
                int ti = wid + u * 8;
                int I = ti >> 2, J = ti & 3;
                int rA = I * 16 + col;          // h1 row of K^T
                v8h a0 = *(const v8h*)&sA[rA * 32 + half * 16];
                v8h a1 = *(const v8h*)&sA[rA * 32 + half * 16 + 8];
                int rB = J * 16 + col;          // h2 row of V^T
                v8h b0 = *(const v8h*)&sB[rB * 32 + half * 16];
                v8h b1 = *(const v8h*)&sB[rB * 32 + half * 16 + 8];
                v16h A, B;
#pragma unroll
                for (int i = 0; i < 8; ++i) { A[i] = a0[i]; A[8 + i] = a1[i]; B[i] = b0[i]; B[8 + i] = b1[i]; }
                accS[m][u] = wmma16(A, B, accS[m][u]);
            }
        }
    }

    // write S matrices to LDS, transposed [h2][h1], as f16
    __syncthreads();
#pragma unroll
    for (int m = 0; m < MODES; ++m)
#pragma unroll
        for (int u = 0; u < 2; ++u) {
            int ti = wid + u * 8;
            int I = ti >> 2, J = ti & 3;
            int h2 = J * 16 + col;
#pragma unroll
            for (int r = 0; r < 8; ++r) {
                int h1 = I * 16 + r + half * 8;
                sS[m][h2 * HDIM + h1] = (_Float16)accS[m][u][r];
            }
        }
    __syncthreads();

    // ---------------- Pass 2: out = sum_m (Q_m/8) @ S_m ----------------
    const int I2 = wid & 1;        // row tile within 32-node chunk
    const int J2 = wid >> 1;       // col tile (0..3)
    for (int c = 0; c < cnt; c += 32) {
        v8f accO;
#pragma unroll
        for (int r = 0; r < 8; ++r) accO[r] = 0.0f;
#pragma unroll
        for (int m = 0; m < MODES; ++m) {
            __syncthreads();
            {   // stage roped, scaled Q chunk [nl][h]
                int n = s + c + nl; bool ok = (n < e);
                float ph = 0.0f;
                if (ok) ph = pos[n * 3] * kvx[m] + pos[n * 3 + 1] * kvy[m] + pos[n * 3 + 2] * kvz[m];
                float cc = __cosf(ph), sn = __sinf(ph);
#pragma unroll
                for (int i = 0; i < 8; ++i) {
                    int h = hb * 8 + i;
                    float q1 = ok ? qs[(size_t)(s + c + nl) * HDIM + h] : 0.0f;
                    float q2 = ok ? qs[(size_t)(s + c + nl) * HDIM + (h ^ 32)] : 0.0f;
                    float val = 0.125f * (q1 * cc + ((h < 32) ? -q2 * sn : q2 * sn));
                    sA[nl * HDIM + h] = (_Float16)val;
                }
            }
            __syncthreads();
#pragma unroll
            for (int kk = 0; kk < 2; ++kk) {   // K = 64 over h1, two 32-steps
                int rQ = I2 * 16 + col;        // node row
                v8h a0 = *(const v8h*)&sA[rQ * HDIM + kk * 32 + half * 16];
                v8h a1 = *(const v8h*)&sA[rQ * HDIM + kk * 32 + half * 16 + 8];
                int h2 = J2 * 16 + col;
                v8h b0 = *(const v8h*)&sS[m][h2 * HDIM + kk * 32 + half * 16];
                v8h b1 = *(const v8h*)&sS[m][h2 * HDIM + kk * 32 + half * 16 + 8];
                v16h A, B;
#pragma unroll
                for (int i = 0; i < 8; ++i) { A[i] = a0[i]; A[8 + i] = a1[i]; B[i] = b0[i]; B[8 + i] = b1[i]; }
                accO = wmma16(A, B, accO);
            }
        }
#pragma unroll
        for (int r = 0; r < 8; ++r) {
            int node = s + c + I2 * 16 + r + half * 8;
            if (node < e) out[(size_t)node * HDIM + J2 * 16 + col] = accO[r];
        }
    }
}

// ---------------------------------------------------------------------------
extern "C" void kernel_launch(void* const* d_in, const int* in_sizes, int n_in,
                              void* d_out, int out_size, void* d_ws, size_t ws_size,
                              hipStream_t stream) {
    const float* nf    = (const float*)d_in[0];  // (N, 64)
    const float* pos   = (const float*)d_in[1];  // (N, 3)
    const float* cell  = (const float*)d_in[2];  // (G, 3, 3)
    const int*   batch = (const int*)  d_in[3];  // (N,)
    const float* W     = (const float*)d_in[4];  // (192, 64)
    const int N = in_sizes[0] / HDIM;
    const int G = in_sizes[2] / 9;

    float* ws   = (float*)d_ws;
    float* qs   = ws;                            // N*64 f32 (silu'd q)
    float* ks   = qs + (size_t)N * HDIM;         // N*64 f32 (silu'd k)
    float* vs   = ks + (size_t)N * HDIM;         // N*64 f32
    float* kvec = vs + (size_t)N * HDIM;         // G*6*3 f32
    int*   rng  = (int*)(kvec + (size_t)G * MODES * 3); // G+1 ints

    fefa_setup_kernel<<<1, 128, 0, stream>>>(cell, batch, N, G, kvec, rng);
    fefa_qkv_kernel<<<(N + 15) / 16, 128, 0, stream>>>(nf, W, qs, ks, vs, N);
    fefa_attn_kernel<<<G, 256, 0, stream>>>(qs, ks, vs, pos, kvec, rng, (float*)d_out);
}
